// CLIPAttention_51977694216622
// MI455X (gfx1250) — compile-verified, matrix operand order checked by reading, not run on hardware
//
#include <hip/hip_runtime.h>

// ---------- CDNA5 types ----------
typedef __attribute__((ext_vector_type(16))) __bf16 v16bf;
typedef __attribute__((ext_vector_type(8)))  __bf16 v8bf;
typedef __attribute__((ext_vector_type(8)))  float  v8f;
typedef __attribute__((ext_vector_type(4)))  int    v4i;

#define NH   12
#define SEQ  1024
#define DIM  768
#define HDIM 64
#define NB   4
#define MTOT (NB * SEQ)   // 4096

#define AS1 __attribute__((address_space(1)))
#define AS3 __attribute__((address_space(3)))

// ---------- async global->LDS staging (CDNA5 GLOBAL_LOAD_ASYNC_TO_LDS_B128) ----------
#if defined(__has_builtin)
#  if __has_builtin(__builtin_amdgcn_global_load_async_to_lds_b128)
#    define HAVE_ASYNC_LDS 1
#  endif
#endif

static __device__ __forceinline__ void cp16_async(void* lds, const void* g) {
#if defined(HAVE_ASYNC_LDS)
    __builtin_amdgcn_global_load_async_to_lds_b128((AS1 v4i*)g, (AS3 v4i*)lds, 0, 0);
#else
    *(uint4*)lds = *(const uint4*)g;
#endif
}

#if defined(HAVE_ASYNC_LDS)
#  if __has_builtin(__builtin_amdgcn_s_wait_asynccnt)
#    define WAIT_ASYNC(n) __builtin_amdgcn_s_wait_asynccnt(n)
#  else
#    define WAIT_ASYNC(n) asm volatile("s_wait_asynccnt %0" :: "i"(n) : "memory")
#  endif
#else
#  define WAIT_ASYNC(n) ((void)0)
#endif

// Fragment loader: 16-bit A/B operand layout for V_WMMA_*_16X16X32.
// Lane l (row/col = l&15) holds K = {b0..b0+7, b0+16..b0+23}, b0 = (l&16)>>1.
static __device__ __forceinline__ v16bf load_frag(const __bf16* rowptr, int lane) {
    int b0 = (lane & 16) >> 1;
    v8bf lo = *(const v8bf*)(rowptr + b0);
    v8bf hi = *(const v8bf*)(rowptr + b0 + 16);
    return __builtin_shufflevector(lo, hi, 0,1,2,3,4,5,6,7,8,9,10,11,12,13,14,15);
}

static __device__ __forceinline__ v8f wmma_bf16(v16bf a, v16bf b, v8f c) {
    return __builtin_amdgcn_wmma_f32_16x16x32_bf16(false, a, false, b, (short)0, c, false, false);
}

// ---------- fp32 -> bf16 conversion ----------
__global__ void cvt_f32_bf16(const float* __restrict__ in, __bf16* __restrict__ out, int n8) {
    int i = blockIdx.x * blockDim.x + threadIdx.x;
    if (i >= n8) return;
    const float4* p = (const float4*)in + (size_t)i * 2;
    float4 x0 = p[0], x1 = p[1];
    v8bf o;
    o[0] = (__bf16)x0.x; o[1] = (__bf16)x0.y; o[2] = (__bf16)x0.z; o[3] = (__bf16)x0.w;
    o[4] = (__bf16)x1.x; o[5] = (__bf16)x1.y; o[6] = (__bf16)x1.z; o[7] = (__bf16)x1.w;
    *((v8bf*)out + i) = o;
}

// ---------- Fused QKV projection: out = X @ W^T + b, bf16 out [B,H,S,HD] ----------
__global__ __launch_bounds__(256) void gemm_qkv(
    const __bf16* __restrict__ Xb,
    const __bf16* __restrict__ Wqb, const __bf16* __restrict__ Wkb, const __bf16* __restrict__ Wvb,
    const float* __restrict__ bq, const float* __restrict__ bk, const float* __restrict__ bv,
    __bf16* __restrict__ Qb, __bf16* __restrict__ Kb, __bf16* __restrict__ Vb)
{
    __shared__ __align__(16) __bf16 As[2][128][40];
    __shared__ __align__(16) __bf16 Bs[2][64][40];
    const int t = threadIdx.x, w = t >> 5, lane = t & 31;
    const int wm = w >> 1, wn = w & 1;
    const int m0 = blockIdx.x * 128, n0 = blockIdx.y * 64;

    const __bf16* W; const float* bias; __bf16* dst; float scale;
    if (blockIdx.z == 0)      { W = Wqb; bias = bq; dst = Qb; scale = 0.125f; }
    else if (blockIdx.z == 1) { W = Wkb; bias = bk; dst = Kb; scale = 1.0f; }
    else                      { W = Wvb; bias = bv; dst = Vb; scale = 1.0f; }

    auto stage = [&](int buf, int k0) {
        #pragma unroll
        for (int i = 0; i < 2; ++i) {
            int idx = t + i * 256;
            int row = idx >> 2, cc = idx & 3;
            cp16_async(&As[buf][row][cc * 8], &Xb[(size_t)(m0 + row) * DIM + k0 + cc * 8]);
        }
        int row = t >> 2, cc = t & 3;
        cp16_async(&Bs[buf][row][cc * 8], &W[(size_t)(n0 + row) * DIM + k0 + cc * 8]);
    };

    v8f zv = {};
    v8f acc[2][2];
    #pragma unroll
    for (int i = 0; i < 2; ++i)
        #pragma unroll
        for (int j = 0; j < 2; ++j) acc[i][j] = zv;

    stage(0, 0);
    int cur = 0;
    for (int k0 = 0; k0 < DIM; k0 += 32) {
        const bool more = (k0 + 32 < DIM);
        if (more) stage(cur ^ 1, k0 + 32);     // async prefetch of next tile into other buffer
        if (more) WAIT_ASYNC(3); else WAIT_ASYNC(0);
        __syncthreads();

        v16bf a0 = load_frag(&As[cur][wm * 32 +  0 + (lane & 15)][0], lane);
        v16bf a1 = load_frag(&As[cur][wm * 32 + 16 + (lane & 15)][0], lane);
        v16bf b0 = load_frag(&Bs[cur][wn * 32 +  0 + (lane & 15)][0], lane);
        v16bf b1 = load_frag(&Bs[cur][wn * 32 + 16 + (lane & 15)][0], lane);
        acc[0][0] = wmma_bf16(a0, b0, acc[0][0]);
        acc[0][1] = wmma_bf16(a0, b1, acc[0][1]);
        acc[1][0] = wmma_bf16(a1, b0, acc[1][0]);
        acc[1][1] = wmma_bf16(a1, b1, acc[1][1]);
        __syncthreads();
        cur ^= 1;
    }

    #pragma unroll
    for (int mt = 0; mt < 2; ++mt)
        #pragma unroll
        for (int nt = 0; nt < 2; ++nt)
            #pragma unroll
            for (int c = 0; c < 8; ++c) {
                int row = m0 + wm * 32 + mt * 16 + c + ((lane & 16) >> 1);
                int col = n0 + wn * 32 + nt * 16 + (lane & 15);
                float v = (acc[mt][nt][c] + bias[col]) * scale;
                int bidx = row >> 10, s = row & 1023;
                int h = col >> 6, dd = col & 63;
                dst[(((size_t)bidx * NH + h) * SEQ + s) * HDIM + dd] = (__bf16)v;
            }
}

// ---------- Causal flash attention: 128 Q-rows per block, 16 rows per wave ----------
__global__ __launch_bounds__(256) void attn(
    const __bf16* __restrict__ Qb, const __bf16* __restrict__ Kb,
    const __bf16* __restrict__ Vb, __bf16* __restrict__ AttnB)
{
    __shared__ __align__(16) __bf16 Ks[64][72];
    __shared__ __align__(16) __bf16 Vts[64][72];     // Vts[d][k] = V[k][d]
    __shared__ __align__(16) __bf16 Ps[8][16][72];   // per-wave P scratch
    const int t = threadIdx.x, w = t >> 5, lane = t & 31;
    const int bh = blockIdx.y;
    const int i0 = blockIdx.x * 128;
    const int r0 = i0 + w * 16;
    const __bf16* Qh = Qb + (size_t)bh * SEQ * HDIM;
    const __bf16* Kh = Kb + (size_t)bh * SEQ * HDIM;
    const __bf16* Vh = Vb + (size_t)bh * SEQ * HDIM;

    const int qrow = r0 + (lane & 15);
    v16bf qf0 = load_frag(Qh + (size_t)qrow * HDIM, lane);
    v16bf qf1 = load_frag(Qh + (size_t)qrow * HDIM + 32, lane);

    v8f zv = {};
    float mst[8], lst[8];
    v8f O[4];
    #pragma unroll
    for (int c = 0; c < 8; ++c) { mst[c] = -3.0e38f; lst[c] = 0.0f; }
    #pragma unroll
    for (int o = 0; o < 4; ++o) O[o] = zv;

    const float NEGC = -4.0525551e18f;   // -(3^39)

    for (int j0 = 0; j0 < i0 + 128; j0 += 64) {
        // K tile: async DMA straight into LDS
        #pragma unroll
        for (int i = 0; i < 2; ++i) {
            int idx = t + i * 256;
            int row = idx >> 3, cc = idx & 7;
            cp16_async(&Ks[row][cc * 8], &Kh[(size_t)(j0 + row) * HDIM + cc * 8]);
        }
        // V tile: transposed into LDS (element scatter, regular path)
        #pragma unroll
        for (int i = 0; i < 2; ++i) {
            int idx = t + i * 256;
            int kr = idx >> 3, cc = idx & 7;
            v8bf vv = *(const v8bf*)&Vh[(size_t)(j0 + kr) * HDIM + cc * 8];
            #pragma unroll
            for (int e = 0; e < 8; ++e) Vts[cc * 8 + e][kr] = vv[e];
        }
        WAIT_ASYNC(0);
        __syncthreads();

        if (j0 <= r0 + 15) {   // wave-uniform causal skip
            // ---- scores S = Q K^T (16x64) ----
            v8f sc[4];
            #pragma unroll
            for (int nt = 0; nt < 4; ++nt) {
                sc[nt] = zv;
                const __bf16* kp = &Ks[nt * 16 + (lane & 15)][0];
                sc[nt] = wmma_bf16(qf0, load_frag(kp, lane), sc[nt]);
                sc[nt] = wmma_bf16(qf1, load_frag(kp + 32, lane), sc[nt]);
            }
            // ---- causal mask + running max ----
            float mnew[8];
            #pragma unroll
            for (int c = 0; c < 8; ++c) mnew[c] = mst[c];
            #pragma unroll
            for (int nt = 0; nt < 4; ++nt) {
                int cn = j0 + nt * 16 + (lane & 15);
                #pragma unroll
                for (int c = 0; c < 8; ++c) {
                    int rm = r0 + c + ((lane & 16) >> 1);
                    float sv = sc[nt][c] + (cn > rm ? NEGC : 0.0f);
                    sc[nt][c] = sv;
                    mnew[c] = fmaxf(mnew[c], sv);
                }
            }
            #pragma unroll
            for (int off = 1; off < 16; off <<= 1)
                #pragma unroll
                for (int c = 0; c < 8; ++c)
                    mnew[c] = fmaxf(mnew[c], __shfl_xor(mnew[c], off, 32));
            // ---- exp / rescale ----
            float alpha[8], rsum[8];
            #pragma unroll
            for (int c = 0; c < 8; ++c) {
                alpha[c] = __expf(mst[c] - mnew[c]);
                mst[c] = mnew[c];
                rsum[c] = 0.0f;
            }
            #pragma unroll
            for (int nt = 0; nt < 4; ++nt)
                #pragma unroll
                for (int c = 0; c < 8; ++c) {
                    float p = __expf(sc[nt][c] - mnew[c]);
                    rsum[c] += p;
                    Ps[w][c + ((lane & 16) >> 1)][nt * 16 + (lane & 15)] = (__bf16)p;
                }
            #pragma unroll
            for (int off = 1; off < 16; off <<= 1)
                #pragma unroll
                for (int c = 0; c < 8; ++c)
                    rsum[c] += __shfl_xor(rsum[c], off, 32);
            #pragma unroll
            for (int c = 0; c < 8; ++c) lst[c] = lst[c] * alpha[c] + rsum[c];

            // ---- O = diag(alpha)*O + P V ----
            const __bf16* pp = &Ps[w][lane & 15][0];
            v16bf pf0 = load_frag(pp, lane);
            v16bf pf1 = load_frag(pp + 32, lane);
            #pragma unroll
            for (int ot = 0; ot < 4; ++ot) {
                #pragma unroll
                for (int c = 0; c < 8; ++c) O[ot][c] *= alpha[c];
                const __bf16* vp = &Vts[ot * 16 + (lane & 15)][0];
                O[ot] = wmma_bf16(pf0, load_frag(vp, lane), O[ot]);
                O[ot] = wmma_bf16(pf1, load_frag(vp + 32, lane), O[ot]);
            }
        }
        __syncthreads();
    }

    const int bidx = bh / NH, h = bh % NH;
    #pragma unroll
    for (int c = 0; c < 8; ++c) {
        float inv = 1.0f / lst[c];
        int srow = r0 + c + ((lane & 16) >> 1);
        #pragma unroll
        for (int ot = 0; ot < 4; ++ot)
            AttnB[((size_t)bidx * SEQ + srow) * DIM + h * HDIM + ot * 16 + (lane & 15)] =
                (__bf16)(O[ot][c] * inv);
    }
}

// ---------- Output projection: out = Attn @ Wo^T + bo (fp32 out) ----------
__global__ __launch_bounds__(256) void gemm_out(
    const __bf16* __restrict__ Ab, const __bf16* __restrict__ Wob,
    const float* __restrict__ bo, float* __restrict__ out)
{
    __shared__ __align__(16) __bf16 As[2][128][40];
    __shared__ __align__(16) __bf16 Bs[2][64][40];
    const int t = threadIdx.x, w = t >> 5, lane = t & 31;
    const int wm = w >> 1, wn = w & 1;
    const int m0 = blockIdx.x * 128, n0 = blockIdx.y * 64;

    auto stage = [&](int buf, int k0) {
        #pragma unroll
        for (int i = 0; i < 2; ++i) {
            int idx = t + i * 256;
            int row = idx >> 2, cc = idx & 3;
            cp16_async(&As[buf][row][cc * 8], &Ab[(size_t)(m0 + row) * DIM + k0 + cc * 8]);
        }
        int row = t >> 2, cc = t & 3;
        cp16_async(&Bs[buf][row][cc * 8], &Wob[(size_t)(n0 + row) * DIM + k0 + cc * 8]);
    };

    v8f zv = {};
    v8f acc[2][2];
    #pragma unroll
    for (int i = 0; i < 2; ++i)
        #pragma unroll
        for (int j = 0; j < 2; ++j) acc[i][j] = zv;

    stage(0, 0);
    int cur = 0;
    for (int k0 = 0; k0 < DIM; k0 += 32) {
        const bool more = (k0 + 32 < DIM);
        if (more) stage(cur ^ 1, k0 + 32);
        if (more) WAIT_ASYNC(3); else WAIT_ASYNC(0);
        __syncthreads();

        v16bf a0 = load_frag(&As[cur][wm * 32 +  0 + (lane & 15)][0], lane);
        v16bf a1 = load_frag(&As[cur][wm * 32 + 16 + (lane & 15)][0], lane);
        v16bf b0 = load_frag(&Bs[cur][wn * 32 +  0 + (lane & 15)][0], lane);
        v16bf b1 = load_frag(&Bs[cur][wn * 32 + 16 + (lane & 15)][0], lane);
        acc[0][0] = wmma_bf16(a0, b0, acc[0][0]);
        acc[0][1] = wmma_bf16(a0, b1, acc[0][1]);
        acc[1][0] = wmma_bf16(a1, b0, acc[1][0]);
        acc[1][1] = wmma_bf16(a1, b1, acc[1][1]);
        __syncthreads();
        cur ^= 1;
    }

    #pragma unroll
    for (int mt = 0; mt < 2; ++mt)
        #pragma unroll
        for (int nt = 0; nt < 2; ++nt)
            #pragma unroll
            for (int c = 0; c < 8; ++c) {
                int row = m0 + wm * 32 + mt * 16 + c + ((lane & 16) >> 1);
                int col = n0 + wn * 32 + nt * 16 + (lane & 15);
                out[(size_t)row * DIM + col] = acc[mt][nt][c] + bo[col];
            }
}

// ---------- host ----------
extern "C" void kernel_launch(void* const* d_in, const int* in_sizes, int n_in,
                              void* d_out, int out_size, void* d_ws, size_t ws_size,
                              hipStream_t stream) {
    const float* X  = (const float*)d_in[0];
    const float* Wq = (const float*)d_in[1]; const float* bq = (const float*)d_in[2];
    const float* Wk = (const float*)d_in[3]; const float* bk = (const float*)d_in[4];
    const float* Wv = (const float*)d_in[5]; const float* bv = (const float*)d_in[6];
    const float* Wo = (const float*)d_in[7]; const float* bo = (const float*)d_in[8];
    float* out = (float*)d_out;

    char* ws = (char*)d_ws;
    const size_t XB_BYTES = (size_t)MTOT * DIM * 2;          // 6 MB
    const size_t WB_BYTES = (size_t)DIM * DIM * 2;           // 1.125 MB
    __bf16* Xb    = (__bf16*)(ws);
    __bf16* Wqb   = (__bf16*)(ws + XB_BYTES);
    __bf16* Wkb   = (__bf16*)(ws + XB_BYTES + WB_BYTES);
    __bf16* Wvb   = (__bf16*)(ws + XB_BYTES + 2 * WB_BYTES);
    __bf16* Wob   = (__bf16*)(ws + XB_BYTES + 3 * WB_BYTES);
    __bf16* Qb    = (__bf16*)(ws + XB_BYTES + 4 * WB_BYTES);
    __bf16* Kb    = (__bf16*)(ws + 2 * XB_BYTES + 4 * WB_BYTES);
    __bf16* Vb    = (__bf16*)(ws + 3 * XB_BYTES + 4 * WB_BYTES);
    __bf16* AttnB = (__bf16*)(ws + 4 * XB_BYTES + 4 * WB_BYTES);

    const int nX8 = MTOT * DIM / 8;       // 393216
    const int nW8 = DIM * DIM / 8;        // 73728
    cvt_f32_bf16<<<nX8 / 256, 256, 0, stream>>>(X,  Xb,  nX8);
    cvt_f32_bf16<<<nW8 / 256, 256, 0, stream>>>(Wq, Wqb, nW8);
    cvt_f32_bf16<<<nW8 / 256, 256, 0, stream>>>(Wk, Wkb, nW8);
    cvt_f32_bf16<<<nW8 / 256, 256, 0, stream>>>(Wv, Wvb, nW8);
    cvt_f32_bf16<<<nW8 / 256, 256, 0, stream>>>(Wo, Wob, nW8);

    gemm_qkv<<<dim3(MTOT / 128, DIM / 64, 3), 256, 0, stream>>>(
        Xb, Wqb, Wkb, Wvb, bq, bk, bv, Qb, Kb, Vb);

    attn<<<dim3(SEQ / 128, NB * NH), 256, 0, stream>>>(Qb, Kb, Vb, AttnB);

    gemm_out<<<dim3(MTOT / 128, DIM / 64), 256, 0, stream>>>(AttnB, Wob, bo, out);
}
